// HarmonicStateSpaceModule_54546084659523
// MI455X (gfx1250) — matrix operand order repaired and checked by reading, not compiled, not run on hardware
//
#include <hip/hip_runtime.h>

typedef __attribute__((ext_vector_type(16))) _Float16 v16h;
typedef __attribute__((ext_vector_type(8)))  float    v8f;

#define BN 8
#define CH 32
#define LL 4096
#define DM 64
#define DI 128
#define DSn 16
#define DRn 4
#define KK 4
#define CPROJ 36   // DR + 2*DS
#define CPAD 48
#define TC 128     // scan LDS chunk (timesteps)

// ---- CDNA5 async-to-LDS path (probe via __has_builtin; fallback = plain LDS store) ----
#if defined(__has_builtin)
#if __has_builtin(__builtin_amdgcn_global_load_async_to_lds_b32) && \
    __has_builtin(__builtin_amdgcn_s_wait_asynccnt)
#define USE_ASYNC_LDS 1
#endif
#endif
#ifndef USE_ASYNC_LDS
#define USE_ASYNC_LDS 0
#endif
#if USE_ASYNC_LDS
typedef __attribute__((address_space(1))) int g1_int;   // global int*
typedef __attribute__((address_space(3))) int l3_int;   // LDS int*
#endif

// ---------------- small math helpers ----------------
__device__ __forceinline__ int brev6(int v) { return (int)(__brev((unsigned)v) >> 26); }
__device__ __forceinline__ float sigmoidf_(float x) { return 1.0f / (1.0f + __expf(-x)); }
__device__ __forceinline__ float siluf_(float x) { return x * sigmoidf_(x); }
__device__ __forceinline__ float softplusf_(float x) { return (x > 20.0f) ? x : log1pf(__expf(x)); }

// direction index map (involution): k bit0 = transpose, bit1 = reverse
__device__ __forceinline__ int xs_index(int k, int l) {
  int lm = (k & 2) ? (4095 - l) : l;
  return (k & 1) ? (((lm & 63) << 6) | (lm >> 6)) : lm;
}

// ---------------- WMMA fragment loaders (wave32, 16x16x32 f16) ----------------
// A (16xK tile, row major in LDS): ISA 7.12.2 "16-bit A-Matrix 16x32" layout.
// Per lane the K indices form two runs of 8 consecutive halves -> 2x ds_load_b128.
__device__ __forceinline__ v16h load_frag_a(const _Float16* A, int lda, int k0) {
  int lane = threadIdx.x & 31;
  int row  = lane & 15;
  int g    = (lane >> 4) & 1;
  v16h out;
#pragma unroll
  for (int e = 0; e < 16; ++e) {
    int v = e >> 1, h = e & 1;
    int k = ((v >= 4) ? 16 : 0) + g * 8 + (v & 3) * 2 + h;
    out[e] = A[(size_t)row * lda + k0 + k];
  }
  return out;
}
// B stored N-major in LDS: Bt[n][k], K contiguous. B layout matches A: lanes 0-15 hold
// K=0..7,16..23 of column n=lane, lanes 16-31 hold K=8..15,24..31. Two runs of 8 halves/lane
// -> 2x ds_load_b128 (no v_mov_b16 packing).
__device__ __forceinline__ v16h load_frag_bT(const _Float16* Bt, int ldk, int k0) {
  int lane = threadIdx.x & 31;
  int col  = lane & 15;
  int g    = (lane >> 4) & 1;
  v16h out;
#pragma unroll
  for (int e = 0; e < 16; ++e)
    out[e] = Bt[(size_t)col * ldk + k0 + (e >> 3) * 16 + g * 8 + (e & 7)];
  return out;
}

// ---------------- 64x64 in-LDS radix-2 FFT (forward, DIT) ----------------
__device__ void fft2d64(float* re, float* im, const float* twr, const float* twi) {
  int tid = threadIdx.x;
  for (int idx = tid; idx < 4096; idx += 256) {
    int r = idx >> 6, c = idx & 63, rb = brev6(c);
    if (c < rb) {
      float t0 = re[r * 64 + c]; re[r * 64 + c] = re[r * 64 + rb]; re[r * 64 + rb] = t0;
      float t1 = im[r * 64 + c]; im[r * 64 + c] = im[r * 64 + rb]; im[r * 64 + rb] = t1;
    }
  }
  __syncthreads();
  for (int s = 1; s <= 6; ++s) {
    int half = 1 << (s - 1);
    for (int t = tid; t < 2048; t += 256) {
      int r = t >> 5, j = t & 31;
      int grp = j >> (s - 1), pos = j & (half - 1);
      int i0 = r * 64 + grp * (half << 1) + pos, i1 = i0 + half;
      int ti = pos << (6 - s);
      float wr = twr[ti], wi = twi[ti];
      float xr1 = re[i1], xi1 = im[i1];
      float tr = wr * xr1 - wi * xi1, tj = wr * xi1 + wi * xr1;
      float ar = re[i0], ai = im[i0];
      re[i0] = ar + tr; im[i0] = ai + tj;
      re[i1] = ar - tr; im[i1] = ai - tj;
    }
    __syncthreads();
  }
  for (int idx = tid; idx < 4096; idx += 256) {
    int r = idx >> 6, c = idx & 63, rb = brev6(r);
    if (r < rb) {
      float t0 = re[r * 64 + c]; re[r * 64 + c] = re[rb * 64 + c]; re[rb * 64 + c] = t0;
      float t1 = im[r * 64 + c]; im[r * 64 + c] = im[rb * 64 + c]; im[rb * 64 + c] = t1;
    }
  }
  __syncthreads();
  for (int s = 1; s <= 6; ++s) {
    int half = 1 << (s - 1);
    for (int t = tid; t < 2048; t += 256) {
      int c = t >> 5, j = t & 31;
      int grp = j >> (s - 1), pos = j & (half - 1);
      int i0 = (grp * (half << 1) + pos) * 64 + c, i1 = i0 + half * 64;
      int ti = pos << (6 - s);
      float wr = twr[ti], wi = twi[ti];
      float xr1 = re[i1], xi1 = im[i1];
      float tr = wr * xr1 - wi * xi1, tj = wr * xi1 + wi * xr1;
      float ar = re[i0], ai = im[i0];
      re[i0] = ar + tr; im[i0] = ai + tj;
      re[i1] = ar - tr; im[i1] = ai - tj;
    }
    __syncthreads();
  }
}

// ---------------- K1: forward FFT + fftshift -> 64 real channels ----------------
__global__ __launch_bounds__(256) void k_fft_fwd(const float* __restrict__ x,
                                                 float* __restrict__ xr) {
  __shared__ float sre[4096], sim[4096];
  __shared__ float twr[32], twi[32];
  int tid = threadIdx.x;
  int b = blockIdx.x >> 5, c = blockIdx.x & 31;
  const float* src = x + (size_t)(b * CH + c) * LL;
  for (int i = tid; i < LL; i += 256) { sre[i] = src[i]; sim[i] = 0.0f; }
  if (tid < 32) {
    float ang = -6.2831853071795864f * (float)tid / 64.0f;
    twr[tid] = __cosf(ang); twi[tid] = __sinf(ang);
  }
  __syncthreads();
  fft2d64(sre, sim, twr, twi);
  float* ore = xr + (size_t)(b * 64 + c) * LL;
  float* oim = xr + (size_t)(b * 64 + c + 32) * LL;
  for (int i = tid; i < LL; i += 256) {
    int h = i >> 6, w = i & 63;
    int s = ((h ^ 32) << 6) | (w ^ 32);   // fftshift for N=64 == index XOR 32
    ore[i] = sre[s];
    oim[i] = sim[s];
  }
}

// ---------------- K2: adjust GEMM (M=32768,N=64,K=64) + bias -> h (b,l,64) ----------------
__global__ __launch_bounds__(128) void k_adj(const float* __restrict__ xr,
                                             const float* __restrict__ adj_w,
                                             const float* __restrict__ adj_b,
                                             float* __restrict__ xa) {
  __shared__ _Float16 As[16][64];
  __shared__ _Float16 Bt[64][64];          // [N][K] == adj_w layout
  int tid = threadIdx.x;
  int m0 = blockIdx.x * 16;
  for (int i = tid; i < 64 * 64; i += 128)
    (&Bt[0][0])[i] = (_Float16)adj_w[i];
  for (int i = tid; i < 16 * 64; i += 128) {
    int r = i >> 6, cc = i & 63;
    int m = m0 + r, b = m >> 12, l = m & 4095;
    As[r][cc] = (_Float16)xr[(size_t)(b * 64 + cc) * LL + l];
  }
  __syncthreads();
  int wv = tid >> 5;
  v8f acc = {};
#pragma unroll
  for (int k0 = 0; k0 < 64; k0 += 32) {
    v16h a = load_frag_a(&As[0][0], 64, k0);
    v16h bf = load_frag_bT(&Bt[wv * 16][0], 64, k0);
    acc = __builtin_amdgcn_wmma_f32_16x16x32_f16(false, a, false, bf, (short)0, acc, false, false);
  }
  int lane = tid & 31, n = lane & 15, mb = (lane >> 4) * 8;
#pragma unroll
  for (int e = 0; e < 8; ++e) {
    int m = m0 + mb + e, col = wv * 16 + n;
    xa[(size_t)m * 64 + col] = acc[e] + adj_b[col];
  }
}

// ---------------- K3: LayerNorm over 64 channels ----------------
__global__ __launch_bounds__(256) void k_ln1(const float* __restrict__ xa,
                                             const float* __restrict__ g,
                                             const float* __restrict__ bt,
                                             float* __restrict__ out) {
  int m = blockIdx.x * 256 + threadIdx.x;
  const float* row = xa + (size_t)m * 64;
  float s = 0.f, s2 = 0.f;
  for (int i = 0; i < 64; ++i) { float v = row[i]; s += v; s2 += v * v; }
  float mean = s * (1.0f / 64.0f);
  float var = s2 * (1.0f / 64.0f) - mean * mean;
  float inv = rsqrtf(var + 1e-5f);
  float* o = out + (size_t)m * 64;
  for (int i = 0; i < 64; ++i) o[i] = (row[i] - mean) * inv * g[i] + bt[i];
}

// ---------------- K4: in-proj GEMM (M=32768,N=256,K=64) -> xp (b,d,l), z (m,d) ----------------
__global__ __launch_bounds__(128) void k_inproj(const float* __restrict__ hln,
                                                const float* __restrict__ w_in,
                                                float* __restrict__ xp,
                                                float* __restrict__ zb) {
  __shared__ _Float16 As[16][64];
  __shared__ _Float16 Bt[64][64];          // rows n0..n0+63 of in_proj_w ([N][K])
  int tid = threadIdx.x;
  int m0 = blockIdx.x * 16;
  int n0 = blockIdx.y * 64;
  for (int i = tid; i < 64 * 64; i += 128)
    (&Bt[0][0])[i] = (_Float16)w_in[(size_t)n0 * 64 + i];
  for (int i = tid; i < 16 * 64; i += 128) {
    int r = i >> 6, cc = i & 63;
    As[r][cc] = (_Float16)hln[(size_t)(m0 + r) * 64 + cc];
  }
  __syncthreads();
  int wv = tid >> 5;
  v8f acc = {};
#pragma unroll
  for (int k0 = 0; k0 < 64; k0 += 32) {
    v16h a = load_frag_a(&As[0][0], 64, k0);
    v16h bf = load_frag_bT(&Bt[wv * 16][0], 64, k0);
    acc = __builtin_amdgcn_wmma_f32_16x16x32_f16(false, a, false, bf, (short)0, acc, false, false);
  }
  int lane = tid & 31, n = lane & 15, mb = (lane >> 4) * 8;
#pragma unroll
  for (int e = 0; e < 8; ++e) {
    int m = m0 + mb + e, j = n0 + wv * 16 + n;
    if (j < DI) {
      int b = m >> 12, l = m & 4095;
      xp[(size_t)(b * DI + j) * LL + l] = acc[e];
    } else {
      zb[(size_t)m * DI + (j - DI)] = acc[e];
    }
  }
}

// ---------------- K5: depthwise 3x3 SAME conv + bias + SiLU ----------------
__global__ __launch_bounds__(256) void k_dwconv(const float* __restrict__ xp,
                                                const float* __restrict__ dww,
                                                const float* __restrict__ dwb,
                                                float* __restrict__ xc) {
  int idx = blockIdx.x * 256 + threadIdx.x;      // b*DI*LL + d*LL + l
  int l = idx & 4095, d = (idx >> 12) & 127, b = idx >> 19;
  int h = l >> 6, w = l & 63;
  const float* in = xp + (size_t)(b * DI + d) * LL;
  float acc = dwb[d];
#pragma unroll
  for (int i = 0; i < 3; ++i) {
    int hy = h + i - 1;
    if (hy < 0 || hy > 63) continue;
#pragma unroll
    for (int j = 0; j < 3; ++j) {
      int wx = w + j - 1;
      if (wx < 0 || wx > 63) continue;
      acc += in[hy * 64 + wx] * dww[d * 9 + i * 3 + j];
    }
  }
  xc[idx] = siluf_(acc);
}

// ---------------- K6: x_proj GEMM per (b,k): M=4096(l), N=36->48, K=128 ----------------
__global__ __launch_bounds__(96) void k_xproj(const float* __restrict__ xc,
                                              const float* __restrict__ xpw,
                                              float* __restrict__ xdbl) {
  __shared__ _Float16 As[16][128];
  __shared__ _Float16 Bt[CPAD][128];       // [N][K], rows >=36 zero
  int tid = threadIdx.x;
  int l0 = blockIdx.x * 16;
  int bk = blockIdx.y, b = bk >> 2, k = bk & 3;
  for (int i = tid; i < CPAD * 128; i += 96) {
    int cc = i >> 7, dd = i & 127;
    Bt[cc][dd] = (cc < CPROJ) ? (_Float16)xpw[(size_t)(k * CPROJ + cc) * DI + dd]
                              : (_Float16)0.0f;
  }
  for (int i = tid; i < 16 * 128; i += 96) {
    int r = i >> 7, dd = i & 127;
    As[r][dd] = (_Float16)xc[(size_t)(b * DI + dd) * LL + xs_index(k, l0 + r)];
  }
  __syncthreads();
  int wv = tid >> 5;                 // 0..2 -> N tiles covering 48 cols
  v8f acc = {};
#pragma unroll
  for (int k0 = 0; k0 < 128; k0 += 32) {
    v16h a = load_frag_a(&As[0][0], 128, k0);
    v16h bf = load_frag_bT(&Bt[wv * 16][0], 128, k0);
    acc = __builtin_amdgcn_wmma_f32_16x16x32_f16(false, a, false, bf, (short)0, acc, false, false);
  }
  int lane = tid & 31, n = lane & 15, mb = (lane >> 4) * 8;
#pragma unroll
  for (int e = 0; e < 8; ++e) {
    int cc = wv * 16 + n;
    if (cc < CPROJ)
      xdbl[((size_t)(b * KK + k) * CPROJ + cc) * LL + (l0 + mb + e)] = acc[e];
  }
}

// ---------------- K7: selective scan, one direction k (launched 4x, serialized) ----------------
// Stages [36][TC] slabs of x_dbl into LDS per chunk via the async-to-LDS path.
__global__ __launch_bounds__(128) void k_scan(const float* __restrict__ xc,
                                              const float* __restrict__ xdbl,
                                              const float* __restrict__ dt_w,
                                              const float* __restrict__ dt_b,
                                              const float* __restrict__ A_logs,
                                              const float* __restrict__ Dsv,
                                              float* __restrict__ yco, int k) {
  __shared__ float cb[CPROJ][TC];
  int d = threadIdx.x;               // 0..127
  int b = blockIdx.x;                // 0..7
  int kd = k * DI + d;
  float Ar[DSn], hst[DSn];
#pragma unroll
  for (int n = 0; n < DSn; ++n) { Ar[n] = -__expf(A_logs[kd * DSn + n]); hst[n] = 0.0f; }
  float wr0 = dt_w[kd * 4 + 0], wr1 = dt_w[kd * 4 + 1];
  float wr2 = dt_w[kd * 4 + 2], wr3 = dt_w[kd * 4 + 3];
  float dtb = dt_b[kd], Dd = Dsv[kd];
  const float* xcb = xc + (size_t)(b * DI + d) * LL;
  const float* xdb = xdbl + (size_t)(b * KK + k) * CPROJ * LL;
  float* yout = yco + (size_t)(b * DI + d) * LL;

  for (int t0 = 0; t0 < LL; t0 += TC) {
    __syncthreads();                 // previous chunk fully consumed
    for (int i = threadIdx.x; i < CPROJ * TC; i += 128) {
      int row = i >> 7, tcc = i & (TC - 1);
      const float* src = xdb + (size_t)row * LL + t0 + tcc;
#if USE_ASYNC_LDS
      __builtin_amdgcn_global_load_async_to_lds_b32(
          (g1_int*)(uintptr_t)src, (l3_int*)(uintptr_t)&cb[row][tcc], 0, 0);
#else
      cb[row][tcc] = *src;
#endif
    }
#if USE_ASYNC_LDS
    __builtin_amdgcn_s_wait_asynccnt(0);
#endif
    __syncthreads();

    for (int tt = 0; tt < TC; ++tt) {
      int t = t0 + tt;
      int p = xs_index(k, t);        // gather index == scatter index (involution)
      float u = xcb[p];
      float delta = softplusf_(cb[0][tt] * wr0 + cb[1][tt] * wr1 +
                               cb[2][tt] * wr2 + cb[3][tt] * wr3 + dtb);
      float du = delta * u;
      float y = 0.0f;
#pragma unroll
      for (int n = 0; n < DSn; ++n) {
        hst[n] = hst[n] * __expf(delta * Ar[n]) + du * cb[DRn + n][tt];
        y += hst[n] * cb[DRn + DSn + n][tt];
      }
      y += Dd * u;
      if (k == 0) yout[p] = y;
      else        yout[p] += y;
    }
  }
}

// ---------------- K8: out-LN over 128 + SiLU(z) gate -> yg (m,d) ----------------
__global__ __launch_bounds__(256) void k_lngate(const float* __restrict__ yco,
                                                const float* __restrict__ zb,
                                                const float* __restrict__ g,
                                                const float* __restrict__ bt,
                                                float* __restrict__ yg) {
  int m = blockIdx.x * 256 + threadIdx.x;
  int b = m >> 12, l = m & 4095;
  const float* base = yco + (size_t)b * DI * LL + l;
  float s = 0.f, s2 = 0.f;
  for (int dd = 0; dd < DI; ++dd) { float v = base[(size_t)dd * LL]; s += v; s2 += v * v; }
  float mean = s * (1.0f / 128.0f);
  float var = s2 * (1.0f / 128.0f) - mean * mean;
  float inv = rsqrtf(var + 1e-5f);
  for (int dd = 0; dd < DI; ++dd) {
    float v = (base[(size_t)dd * LL] - mean) * inv * g[dd] + bt[dd];
    yg[(size_t)m * DI + dd] = v * siluf_(zb[(size_t)m * DI + dd]);
  }
}

// ---------------- K9: out-proj GEMM (M=32768,N=64,K=128) + residual into xa ----------------
__global__ __launch_bounds__(128) void k_outproj(const float* __restrict__ yg,
                                                 const float* __restrict__ w_out,
                                                 float* __restrict__ xa) {
  __shared__ _Float16 As[16][128];
  __shared__ _Float16 Bt[64][128];         // [N][K] == out_proj_w layout
  int tid = threadIdx.x;
  int m0 = blockIdx.x * 16;
  for (int i = tid; i < 64 * 128; i += 128)
    (&Bt[0][0])[i] = (_Float16)w_out[i];
  for (int i = tid; i < 16 * 128; i += 128) {
    int r = i >> 7, dd = i & 127;
    As[r][dd] = (_Float16)yg[(size_t)(m0 + r) * DI + dd];
  }
  __syncthreads();
  int wv = tid >> 5;
  v8f acc = {};
#pragma unroll
  for (int k0 = 0; k0 < 128; k0 += 32) {
    v16h a = load_frag_a(&As[0][0], 128, k0);
    v16h bf = load_frag_bT(&Bt[wv * 16][0], 128, k0);
    acc = __builtin_amdgcn_wmma_f32_16x16x32_f16(false, a, false, bf, (short)0, acc, false, false);
  }
  int lane = tid & 31, n = lane & 15, mb = (lane >> 4) * 8;
#pragma unroll
  for (int e = 0; e < 8; ++e) {
    int m = m0 + mb + e, col = wv * 16 + n;
    xa[(size_t)m * 64 + col] += acc[e];          // residual add
  }
}

// ---------------- K10: ifftshift + inverse FFT (conjugate trick) -> real out ----------------
__global__ __launch_bounds__(256) void k_ifft(const float* __restrict__ xa,
                                              float* __restrict__ out) {
  __shared__ float sre[4096], sim[4096];
  __shared__ float twr[32], twi[32];
  int tid = threadIdx.x;
  int b = blockIdx.x >> 5, c = blockIdx.x & 31;
  for (int i = tid; i < LL; i += 256) {
    int h = i >> 6, w = i & 63;
    int lsrc = ((h ^ 32) << 6) | (w ^ 32);       // ifftshift for N=64
    size_t base = ((size_t)b * LL + (size_t)lsrc) * 64;
    sre[i] = xa[base + c];
    sim[i] = -xa[base + c + 32];                 // conjugate
  }
  if (tid < 32) {
    float ang = -6.2831853071795864f * (float)tid / 64.0f;
    twr[tid] = __cosf(ang); twi[tid] = __sinf(ang);
  }
  __syncthreads();
  fft2d64(sre, sim, twr, twi);
  float* o = out + (size_t)(b * CH + c) * LL;
  for (int i = tid; i < LL; i += 256)
    o[i] = sre[i] * (1.0f / 4096.0f);            // Re(conj(z)) == Re(z)
}

// ---------------- launch ----------------
extern "C" void kernel_launch(void* const* d_in, const int* in_sizes, int n_in,
                              void* d_out, int out_size, void* d_ws, size_t ws_size,
                              hipStream_t stream) {
  (void)in_sizes; (void)n_in; (void)out_size; (void)ws_size;
  const float* x        = (const float*)d_in[0];
  const float* adj_w    = (const float*)d_in[1];
  const float* adj_b    = (const float*)d_in[2];
  const float* ln1_g    = (const float*)d_in[3];
  const float* ln1_b    = (const float*)d_in[4];
  const float* in_proj  = (const float*)d_in[5];
  const float* dw_w     = (const float*)d_in[6];
  const float* dw_b     = (const float*)d_in[7];
  const float* x_proj_w = (const float*)d_in[8];
  const float* dt_w     = (const float*)d_in[9];
  const float* dt_b     = (const float*)d_in[10];
  const float* A_logs   = (const float*)d_in[11];
  const float* Ds       = (const float*)d_in[12];
  const float* oln_g    = (const float*)d_in[13];
  const float* oln_b    = (const float*)d_in[14];
  const float* out_proj = (const float*)d_in[15];
  float* out = (float*)d_out;

  float* ws   = (float*)d_ws;
  float* xr   = ws;                              // (b,64,l)
  float* xa   = xr   + (size_t)BN * 64 * LL;     // (b,l,64)
  float* hln  = xa   + (size_t)BN * LL * 64;
  float* xp   = hln  + (size_t)BN * LL * 64;     // (b,d,l)
  float* zb   = xp   + (size_t)BN * DI * LL;     // (m,d)
  float* xc   = zb   + (size_t)BN * LL * DI;     // (b,d,l)
  float* xdbl = xc   + (size_t)BN * DI * LL;     // (b,k,36,l)
  float* yco  = xdbl + (size_t)BN * KK * CPROJ * LL;   // (b,d,l)
  float* yg   = xp;                              // reuse xp region after conv

  k_fft_fwd<<<BN * CH, 256, 0, stream>>>(x, xr);
  k_adj<<<(BN * LL) / 16, 128, 0, stream>>>(xr, adj_w, adj_b, xa);
  k_ln1<<<(BN * LL) / 256, 256, 0, stream>>>(xa, ln1_g, ln1_b, hln);
  k_inproj<<<dim3((BN * LL) / 16, 4), 128, 0, stream>>>(hln, in_proj, xp, zb);
  k_dwconv<<<(BN * DI * LL) / 256, 256, 0, stream>>>(xp, dw_w, dw_b, xc);
  k_xproj<<<dim3(LL / 16, BN * KK), 96, 0, stream>>>(xc, x_proj_w, xdbl);
  for (int k = 0; k < KK; ++k)
    k_scan<<<BN, DI, 0, stream>>>(xc, xdbl, dt_w, dt_b, A_logs, Ds, yco, k);
  k_lngate<<<(BN * LL) / 256, 256, 0, stream>>>(yco, zb, oln_g, oln_b, yg);
  k_outproj<<<(BN * LL) / 16, 128, 0, stream>>>(yg, out_proj, xa);
  k_ifft<<<BN * CH, 256, 0, stream>>>(xa, out);
}
